// scTransNet_GAT_26611617366516
// MI455X (gfx1250) — matrix operand order repaired and checked by reading, not compiled, not run on hardware
//
#include <hip/hip_runtime.h>
#include <hip/hip_bf16.h>

typedef __attribute__((ext_vector_type(16))) __bf16 v16bf;
typedef __attribute__((ext_vector_type(8)))  float  v8f;

#define N_NODES 8192
#define DIN     512
#define DH      256
#define NHEAD   2
#define DG      512
#define M1DIM   512
#define M2DIM   256
#define ALPHA_GAT 0.2f
#define MLP_SLOPE 0.01f
#define LDT     40     // padded K-stride (elements) for LDS tiles: 80B, 16B-aligned
#define NEG_BIG (-1.0e30f)

// ---------- small helpers ----------
static __device__ __forceinline__ unsigned short f32_to_bf16(float f) {
    union { float f; unsigned u; } c; c.f = f;
    unsigned r = c.u + 0x7FFFu + ((c.u >> 16) & 1u);   // round-nearest-even
    return (unsigned short)(r >> 16);
}
static __device__ __forceinline__ float bf16_to_f32(unsigned short u) {
    union { unsigned u; float f; } c; c.u = ((unsigned)u) << 16;
    return c.f;
}
static __device__ __forceinline__ float lrelu(float v, float slope) {
    return v > 0.0f ? v : slope * v;
}
static __device__ __forceinline__ float fast_rcp(float v) {
    return __builtin_amdgcn_rcpf(v);
}

// ---------- WMMA fragment loaders (CDNA5 16x16x32 bf16 lane layouts) ----------
// A tile in LDS: row-major [M][LDT] (K contiguous per row).
// A (16x32, MxK): lanes 0-15 row M=lane hold K {0..7,16..23}; lanes 16-31 hold K {8..15,24..31}
static __device__ __forceinline__ v16bf load_A_frag(const unsigned short* Ab, int mt, int lane) {
    int row = mt * 16 + (lane & 15);
    int kb  = (lane >> 4) * 8;
    union { uint4 q[2]; v16bf v; } u;
    u.q[0] = *reinterpret_cast<const uint4*>(Ab + row * LDT + kb);        // K = kb..kb+7
    u.q[1] = *reinterpret_cast<const uint4*>(Ab + row * LDT + 16 + kb);   // K = 16+kb..
    return u.v;
}
// B tile in LDS: TRANSPOSED [N][LDT] (K contiguous per column of B).
// B (32x16, KxN): lanes 0-15 col N=lane hold K {0..7,16..23}; lanes 16-31 hold K {8..15,24..31}
static __device__ __forceinline__ v16bf load_B_frag_t(const unsigned short* Bt, int nt, int lane) {
    int col = nt * 16 + (lane & 15);
    int kg  = (lane >> 4) * 8;
    union { uint4 q[2]; v16bf v; } u;
    const unsigned short* p = Bt + col * LDT + kg;
    u.q[0] = *reinterpret_cast<const uint4*>(p);        // K = kg..kg+7
    u.q[1] = *reinterpret_cast<const uint4*>(p + 16);   // K = 16+kg..16+kg+7
    return u.v;
}
static __device__ __forceinline__ v8f wmma_bf16(v16bf a, v16bf b, v8f c) {
    return __builtin_amdgcn_wmma_f32_16x16x32_bf16(false, a, false, b, (short)0, c, false, false);
}
// convert 8 f32 -> one 16B packet of bf16
static __device__ __forceinline__ uint4 pack_bf16_8(const float* f) {
    union { unsigned short s[8]; uint4 q; } u;
#pragma unroll
    for (int i = 0; i < 8; ++i) u.s[i] = f32_to_bf16(f[i]);
    return u.q;
}

// ---------- generic bf16 WMMA GEMM: C = A(MxK) * B(KxN), 128x64 block tile ----------
// Double-buffered LDS, B-fragment prefetch pipeline.
// MODE 0: A=f32, plain store bf16 (x@Wg -> h)
// MODE 1: A=bf16, epilogue bias+lrelu(0.01), store bf16 (embed@W1 -> z1)
// MODE 2: A=bf16, epilogue bias+lrelu(0.01), store f32  (z1@W2 -> tf)
template <int MODE>
__global__ __launch_bounds__(256) void gemm_bf16_kernel(
    const void* __restrict__ Asrc, const float* __restrict__ Bsrc,
    const float* __restrict__ bias, void* __restrict__ Cdst,
    int K, int Nld, long strideBH, long strideCH)
{
    constexpr int BM = 128, BK = 32;
    constexpr int ASZ = BM * LDT, BSZ = 64 * LDT;
    __shared__ alignas(16) unsigned short AbS[2 * ASZ];   // [M][K] x2
    __shared__ alignas(16) unsigned short BtS[2 * BSZ];   // [N][K] (transposed) x2

    const int tid  = threadIdx.x;
    const int lane = tid & 31;
    const int w    = tid >> 5;           // wave id = M tile
    const int hd   = blockIdx.z;
    const int m0   = blockIdx.x * BM;
    const int n0   = blockIdx.y * 64;
    const float* Bf = Bsrc + (long)hd * strideBH;

    // staging coords
    const int arow = tid >> 1, ac0 = (tid & 1) * 16;   // A: 16 elems per thread
    const int brow = tid >> 3, bc0 = (tid & 7) * 8;    // B: 8 elems per thread

    // staged registers
    float af[16]; uint4 au[2]; float bfv[8];

    auto global_load_tile = [&](int k0) {
        if constexpr (MODE == 0) {
            const float* s = (const float*)Asrc + (long)(m0 + arow) * K + k0 + ac0;
#pragma unroll
            for (int u = 0; u < 16; ++u) af[u] = s[u];
        } else {
            const uint4* s = reinterpret_cast<const uint4*>(
                (const unsigned short*)Asrc + (long)(m0 + arow) * K + k0 + ac0);
            au[0] = s[0]; au[1] = s[1];
        }
        const float* bs = Bf + (long)(k0 + brow) * Nld + n0 + bc0;
#pragma unroll
        for (int u = 0; u < 8; ++u) bfv[u] = bs[u];
    };
    auto lds_store_tile = [&](int buf) {
        unsigned short* ad = AbS + buf * ASZ + arow * LDT + ac0;
        if constexpr (MODE == 0) {
            *reinterpret_cast<uint4*>(ad)     = pack_bf16_8(af);
            *reinterpret_cast<uint4*>(ad + 8) = pack_bf16_8(af + 8);
        } else {
            *reinterpret_cast<uint4*>(ad)     = au[0];
            *reinterpret_cast<uint4*>(ad + 8) = au[1];
        }
        unsigned short* bd = BtS + buf * BSZ;
#pragma unroll
        for (int u = 0; u < 8; ++u)
            bd[(bc0 + u) * LDT + brow] = f32_to_bf16(bfv[u]);
    };

    v8f acc[4] = {};
    const int nk = K / BK;

    global_load_tile(0);
    lds_store_tile(0);
    __syncthreads();

    for (int ki = 0; ki < nk; ++ki) {
        const int buf = ki & 1;
        if (ki + 1 < nk) global_load_tile((ki + 1) * BK);   // prefetch next chunk

        const unsigned short* Ab = AbS + buf * ASZ;
        const unsigned short* Bt = BtS + buf * BSZ;
        v16bf a    = load_A_frag(Ab, w, lane);
        v16bf bcur = load_B_frag_t(Bt, 0, lane);
#pragma unroll
        for (int nt = 0; nt < 4; ++nt) {
            v16bf bnext;
            if (nt < 3) bnext = load_B_frag_t(Bt, nt + 1, lane);
            acc[nt] = wmma_bf16(a, bcur, acc[nt]);
            bcur = bnext;
        }

        if (ki + 1 < nk) lds_store_tile((ki + 1) & 1);
        __syncthreads();
    }

    const int lg = lane >> 4, cl = lane & 15;
#pragma unroll
    for (int nt = 0; nt < 4; ++nt) {
#pragma unroll
        for (int r = 0; r < 8; ++r) {
            int row = m0 + w * 16 + lg * 8 + r;
            int col = n0 + nt * 16 + cl;
            float v = acc[nt][r];
            if constexpr (MODE == 0) {
                ((unsigned short*)Cdst)[(long)hd * strideCH + (long)row * Nld + col] = f32_to_bf16(v);
            } else if constexpr (MODE == 1) {
                v = lrelu(v + bias[col], MLP_SLOPE);
                ((unsigned short*)Cdst)[(long)row * Nld + col] = f32_to_bf16(v);
            } else {
                v = lrelu(v + bias[col], MLP_SLOPE);
                ((float*)Cdst)[(long)row * Nld + col] = v;
            }
        }
    }
}

// ---------- s1 = h@a1, s2 = h@a2 (one wave per row, both heads flattened) ----------
__global__ __launch_bounds__(256) void s1s2_kernel(
    const unsigned short* __restrict__ hmat, const float* __restrict__ a1,
    const float* __restrict__ a2, float* __restrict__ s1, float* __restrict__ s2)
{
    int g    = blockIdx.x * 8 + (threadIdx.x >> 5);   // 0 .. H*N-1
    int lane = threadIdx.x & 31;
    int hd   = g >> 13;
    int i    = g & (N_NODES - 1);
    const unsigned short* hr = hmat + ((long)hd * N_NODES + i) * DH;
    float p1 = 0.0f, p2 = 0.0f;
    for (int c = lane; c < DH; c += 32) {
        float hv = bf16_to_f32(hr[c]);
        p1 += hv * a1[hd * DH + c];
        p2 += hv * a2[hd * DH + c];
    }
#pragma unroll
    for (int off = 16; off > 0; off >>= 1) {
        p1 += __shfl_xor(p1, off, 32);
        p2 += __shfl_xor(p2, off, 32);
    }
    if (lane == 0) { s1[g] = p1; s2[g] = p2; }
}

// ---------- per-row masked online softmax stats: m[i], l[i] per head ----------
// Branchless: finite -1e30 sentinel, mask folded as cndmask/multiply.
__global__ __launch_bounds__(256) void row_stats_kernel(
    const int* __restrict__ adj, const float* __restrict__ s1,
    const float* __restrict__ s2, float* __restrict__ mrow, float* __restrict__ lrow)
{
    const int i   = blockIdx.x;
    const int tid = threadIdx.x;
    float mx[2] = { NEG_BIG, NEG_BIG };
    float sm[2] = { 0.0f, 0.0f };
    float s1v[2] = { s1[i], s1[N_NODES + i] };

    for (int c = tid; c < N_NODES; c += 256) {
        int   av  = adj[(long)i * N_NODES + c];
        float msk = (av > 0) ? 1.0f : 0.0f;
#pragma unroll
        for (int hd = 0; hd < 2; ++hd) {
            float e  = lrelu(s1v[hd] + s2[hd * N_NODES + c], ALPHA_GAT);
            float em = (av > 0) ? e : NEG_BIG;
            float nm = fmaxf(mx[hd], em);
            sm[hd] = sm[hd] * __expf(mx[hd] - nm) + msk * __expf(em - nm);
            mx[hd] = nm;
        }
    }
    __shared__ float smx[2][256], ssm[2][256];
    smx[0][tid] = mx[0]; smx[1][tid] = mx[1];
    ssm[0][tid] = sm[0]; ssm[1][tid] = sm[1];
    __syncthreads();
    for (int off = 128; off > 0; off >>= 1) {
        if (tid < off) {
#pragma unroll
            for (int hd = 0; hd < 2; ++hd) {
                float m1 = smx[hd][tid],     q1 = ssm[hd][tid];
                float m2 = smx[hd][tid+off], q2 = ssm[hd][tid+off];
                float M = fmaxf(m1, m2);
                smx[hd][tid] = M;
                ssm[hd][tid] = q1 * __expf(m1 - M) + q2 * __expf(m2 - M);
            }
        }
        __syncthreads();
    }
    if (tid == 0) {
#pragma unroll
        for (int hd = 0; hd < 2; ++hd) {
            mrow[hd * N_NODES + i] = smx[hd][0];
            lrow[hd * N_NODES + i] = ssm[hd][0];
        }
    }
}

// ---------- fused attention GEMM: out = lrelu(att@h)/||.|| + bias ----------
// Block: 64 rows x all 256 cols; K = 8192 in chunks of 32.
// Double-buffered LDS; att tile built branchlessly on the fly; B-frag prefetch.
__global__ __launch_bounds__(256) void attn_out_kernel(
    const int* __restrict__ adj, const unsigned short* __restrict__ hmat,
    const float* __restrict__ s1, const float* __restrict__ s2,
    const float* __restrict__ mrow, const float* __restrict__ lrow,
    const float* __restrict__ bias_g, float* __restrict__ o0, float* __restrict__ o1)
{
    constexpr int BM = 64, BK = 32;
    constexpr int ASZ = BM * LDT, BSZ = DH * LDT;
    __shared__ alignas(16) unsigned short AbS[2 * ASZ];    // att tile [M][K] x2
    __shared__ alignas(16) unsigned short BtS[2 * BSZ];    // h tile [N][K] (transposed) x2
    __shared__ float ssq[2][BM];

    const int tid  = threadIdx.x;
    const int lane = tid & 31;
    const int w    = tid >> 5;
    const int mt   = w >> 1;            // 4 M tiles
    const int ntb  = (w & 1) * 8;       // 16 N tiles split over wave pairs
    const int hd   = blockIdx.z;
    const int i0   = blockIdx.x * BM;

    // att-tile generator coords (fixed row per thread across the K loop)
    const int ar  = tid >> 2;           // 0..63
    const int ac0 = (tid & 3) * 8;      // 0..24
    const float s1i = s1[hd * N_NODES + i0 + ar];
    const float mi  = mrow[hd * N_NODES + i0 + ar];
    const int*   adjrow = adj + (long)(i0 + ar) * N_NODES + ac0;
    const float* s2base = s2 + hd * N_NODES + ac0;

    // h-tile staging coords
    const int brow = tid >> 3;          // k index 0..31
    const int bc0  = (tid & 7) * 32;    // col chunk
    const unsigned short* hbase = hmat + ((long)hd * N_NODES + brow) * DH + bc0;

    // staged registers
    int aj[8]; float s2v[8];
    union { uint4 q[4]; unsigned short s[32]; } tv;

    auto global_load_tile = [&](int k0) {
#pragma unroll
        for (int u = 0; u < 8; ++u) { aj[u] = adjrow[k0 + u]; s2v[u] = s2base[k0 + u]; }
        const uint4* src = reinterpret_cast<const uint4*>(hbase + (long)k0 * DH);
#pragma unroll
        for (int u = 0; u < 4; ++u) tv.q[u] = src[u];
    };
    auto lds_store_tile = [&](int buf) {
        float pv[8];
#pragma unroll
        for (int u = 0; u < 8; ++u) {
            float msk = (aj[u] > 0) ? 1.0f : 0.0f;          // branchless mask
            float e   = lrelu(s1i + s2v[u], ALPHA_GAT);
            pv[u]     = msk * __expf(e - mi);
        }
        *reinterpret_cast<uint4*>(AbS + buf * ASZ + ar * LDT + ac0) = pack_bf16_8(pv);
        unsigned short* bd = BtS + buf * BSZ;
#pragma unroll
        for (int u = 0; u < 32; ++u)
            bd[(bc0 + u) * LDT + brow] = tv.s[u];
    };

    v8f acc[8] = {};
    const int nk = N_NODES / BK;

    global_load_tile(0);
    lds_store_tile(0);
    __syncthreads();

    for (int ki = 0; ki < nk; ++ki) {
        const int buf = ki & 1;
        if (ki + 1 < nk) global_load_tile((ki + 1) * BK);   // prefetch next chunk

        const unsigned short* Ab = AbS + buf * ASZ;
        const unsigned short* Bt = BtS + buf * BSZ;
        v16bf a    = load_A_frag(Ab, mt, lane);
        v16bf bcur = load_B_frag_t(Bt, ntb, lane);
#pragma unroll
        for (int nt = 0; nt < 8; ++nt) {
            v16bf bnext;
            if (nt < 7) bnext = load_B_frag_t(Bt, ntb + nt + 1, lane);
            acc[nt] = wmma_bf16(a, bcur, acc[nt]);
            bcur = bnext;
        }

        if (ki + 1 < nk) lds_store_tile((ki + 1) & 1);
        __syncthreads();
    }

    // epilogue: divide by l, lrelu(0.2), row l2-normalize, add bias
    const int lg = lane >> 4, cl = lane & 15;
    float invl[8], part[8];
#pragma unroll
    for (int r = 0; r < 8; ++r) {
        int row = i0 + mt * 16 + lg * 8 + r;
        invl[r] = fast_rcp(fmaxf(lrow[hd * N_NODES + row], 1e-30f));
        part[r] = 0.0f;
    }
#pragma unroll
    for (int nt = 0; nt < 8; ++nt)
#pragma unroll
        for (int r = 0; r < 8; ++r) {
            float v = lrelu(acc[nt][r] * invl[r], ALPHA_GAT);
            acc[nt][r] = v;
            part[r] += v * v;
        }
#pragma unroll
    for (int r = 0; r < 8; ++r)
#pragma unroll
        for (int off = 1; off < 16; off <<= 1)
            part[r] += __shfl_xor(part[r], off, 32);
    if (cl == 0)
#pragma unroll
        for (int r = 0; r < 8; ++r)
            ssq[w & 1][mt * 16 + lg * 8 + r] = part[r];
    __syncthreads();

    float* out = (hd == 0) ? o0 : o1;
#pragma unroll
    for (int r = 0; r < 8; ++r) {
        int rl  = mt * 16 + lg * 8 + r;
        float rn = fast_rcp(fmaxf(sqrtf(ssq[0][rl] + ssq[1][rl]), 1e-12f));
#pragma unroll
        for (int nt = 0; nt < 8; ++nt) {
            int col = (ntb + nt) * 16 + cl;
            out[(long)(i0 + rl) * DH + col] = acc[nt][r] * rn + bias_g[hd * DH + col];
        }
    }
}

// ---------- embed = concat(llm_emb, mean(out_heads)) as bf16 ----------
__global__ __launch_bounds__(256) void embed_kernel(
    const float* __restrict__ llm, const float* __restrict__ o0,
    const float* __restrict__ o1, unsigned short* __restrict__ embed)
{
    long idx = (long)blockIdx.x * 256 + threadIdx.x;
    const long total = (long)N_NODES * (DG + DH);
    if (idx >= total) return;
    int row = (int)(idx / (DG + DH));
    int col = (int)(idx % (DG + DH));
    float v = (col < DG) ? llm[(long)row * DG + col]
                         : 0.5f * (o0[(long)row * DH + col - DG] + o1[(long)row * DH + col - DG]);
    embed[idx] = f32_to_bf16(v);
}

// ---------- pred[e] = dot(tf[src], tf[dst]) ----------
__global__ __launch_bounds__(256) void pred_kernel(
    const int* __restrict__ ts, const float* __restrict__ tf,
    float* __restrict__ out, int E)
{
    int e    = blockIdx.x * 8 + (threadIdx.x >> 5);
    int lane = threadIdx.x & 31;
    if (e >= E) return;
    int i = ts[2 * e], j = ts[2 * e + 1];
    const float* a = tf + (long)i * M2DIM;
    const float* b = tf + (long)j * M2DIM;
    float s = 0.0f;
    for (int c = lane; c < M2DIM; c += 32) s += a[c] * b[c];
#pragma unroll
    for (int off = 16; off > 0; off >>= 1) s += __shfl_xor(s, off, 32);
    if (lane == 0) out[e] = s;
}

extern "C" void kernel_launch(void* const* d_in, const int* in_sizes, int n_in,
                              void* d_out, int out_size, void* d_ws, size_t ws_size,
                              hipStream_t stream) {
    const float* x     = (const float*)d_in[0];
    const int*   adj   = (const int*)d_in[1];
    const int*   tsamp = (const int*)d_in[2];
    const float* llm   = (const float*)d_in[3];
    const float* Wg    = (const float*)d_in[4];
    const float* a1    = (const float*)d_in[5];
    const float* a2    = (const float*)d_in[6];
    const float* biasg = (const float*)d_in[7];
    const float* W1    = (const float*)d_in[8];
    const float* b1    = (const float*)d_in[9];
    const float* W2    = (const float*)d_in[10];
    const float* b2    = (const float*)d_in[11];
    const int E = in_sizes[2] / 2;

    // workspace carve-out
    char* ws = (char*)d_ws;
    auto carve = [&](size_t bytes) { void* p = ws; ws += (bytes + 255) & ~(size_t)255; return p; };
    unsigned short* h_bf16 = (unsigned short*)carve((size_t)NHEAD * N_NODES * DH * 2);
    float* s1 = (float*)carve((size_t)NHEAD * N_NODES * 4);
    float* s2 = (float*)carve((size_t)NHEAD * N_NODES * 4);
    float* mr = (float*)carve((size_t)NHEAD * N_NODES * 4);
    float* lr = (float*)carve((size_t)NHEAD * N_NODES * 4);
    float* o0 = (float*)carve((size_t)N_NODES * DH * 4);
    float* o1 = (float*)carve((size_t)N_NODES * DH * 4);
    unsigned short* embed = (unsigned short*)carve((size_t)N_NODES * (DG + DH) * 2);
    unsigned short* z1    = (unsigned short*)carve((size_t)N_NODES * M1DIM * 2);
    float* tf = (float*)carve((size_t)N_NODES * M2DIM * 4);

    // 1) h = x @ Wg[head]   (8192x512x256, bf16 WMMA)
    gemm_bf16_kernel<0><<<dim3(N_NODES / 128, DH / 64, NHEAD), 256, 0, stream>>>(
        x, Wg, nullptr, h_bf16, DIN, DH, (long)DIN * DH, (long)N_NODES * DH);
    // 2) s1, s2
    s1s2_kernel<<<(NHEAD * N_NODES) / 8, 256, 0, stream>>>(h_bf16, a1, a2, s1, s2);
    // 3) masked softmax row stats (one adj pass, both heads)
    row_stats_kernel<<<N_NODES, 256, 0, stream>>>(adj, s1, s2, mr, lr);
    // 4) fused att@h + lrelu + l2norm + bias (one adj pass per head, bf16 WMMA)
    attn_out_kernel<<<dim3(N_NODES / 64, 1, NHEAD), 256, 0, stream>>>(
        adj, h_bf16, s1, s2, mr, lr, biasg, o0, o1);
    // 5) embed = concat(llm_emb, mean(heads)) as bf16
    embed_kernel<<<(N_NODES * (DG + DH)) / 256, 256, 0, stream>>>(llm, o0, o1, embed);
    // 6) z1 = lrelu(embed @ W1 + b1)   (8192x768x512)
    gemm_bf16_kernel<1><<<dim3(N_NODES / 128, M1DIM / 64, 1), 256, 0, stream>>>(
        embed, W1, b1, z1, DG + DH, M1DIM, 0, 0);
    // 7) tf = lrelu(z1 @ W2 + b2)      (8192x512x256) ; tf_embed == target_embed
    gemm_bf16_kernel<2><<<dim3(N_NODES / 128, M2DIM / 64, 1), 256, 0, stream>>>(
        z1, W2, b2, tf, M1DIM, M2DIM, 0, 0);
    // 8) pred
    pred_kernel<<<(E + 7) / 8, 256, 0, stream>>>(tsamp, tf, (float*)d_out, E);
}
